// TFEncoderLayer_88192858456347
// MI455X (gfx1250) — matrix operand order repaired and checked by reading, not compiled, not run on hardware
//
#include <hip/hip_runtime.h>
#include <hip/hip_bf16.h>
#include <math.h>

// ---------------------------------------------------------------- constants
#define BSZ   4
#define DCH   512
#define LSEQ  2048
#define HHEAD 8
#define DKH   64
#define HIDF  2048

typedef __attribute__((ext_vector_type(16))) __bf16 v16bf;
typedef __attribute__((ext_vector_type(8)))  float  v8f;

__device__ __forceinline__ v8f wmma_bf16(v16bf a, v16bf b, v8f c) {
    // D = A(16x32 bf16) * B(32x16 bf16) + C(16x16 f32)
    return __builtin_amdgcn_wmma_f32_16x16x32_bf16(
        false, a, false, b, (short)0, c, false, false);
}

union AFrag { uint4 u[2]; __bf16 h[16]; v16bf v; };
union Pack8 { __bf16 h[8]; uint4 u; };

// ---------------------------------------------------------------- fp32 -> bf16 (weights)
__global__ void cvt_bf16_kernel(const float* __restrict__ in,
                                __bf16* __restrict__ out, size_t n) {
    for (size_t i = (size_t)blockIdx.x * blockDim.x + threadIdx.x; i < n;
         i += (size_t)gridDim.x * blockDim.x)
        out[i] = (__bf16)in[i];
}

// ---------------------------------------------------------------- transpose [D,L] -> [L,D] (+bf16)
// block (32,8); grid (L/32, D/32, B). LDS-tiled, coalesced both sides.
__global__ __launch_bounds__(256)
void transpose_cvt_kernel(const float* __restrict__ in,   // [B, D, L]
                          float* __restrict__ outT,       // [B, L, D]
                          __bf16* __restrict__ outTb) {   // [B, L, D]
    __shared__ float tile[32][33];
    const int b  = blockIdx.z;
    const int c0 = blockIdx.y * 32;
    const int l0 = blockIdx.x * 32;
    const float* src = in + (size_t)b * DCH * LSEQ;
    #pragma unroll
    for (int i = threadIdx.y; i < 32; i += 8)
        tile[i][threadIdx.x] = src[(size_t)(c0 + i) * LSEQ + l0 + threadIdx.x];
    __syncthreads();
    float*  dF = outT  + (size_t)b * LSEQ * DCH;
    __bf16* dB = outTb + (size_t)b * LSEQ * DCH;
    #pragma unroll
    for (int i = threadIdx.y; i < 32; i += 8) {
        const float v = tile[threadIdx.x][i];
        const size_t o = (size_t)(l0 + i) * DCH + c0 + threadIdx.x;
        dF[o] = v;
        dB[o] = (__bf16)v;
    }
}

// ---------------------------------------------------------------- generic GEMM
// C[b] = W[M,Kd] * act[b]  with act stored [N, Kd] (K contiguous per column).
// block = 256 threads = 8 waves (2M x 4N); wave tile 32(M) x 64(N); block 64 x 256.
// All WMMA operand loads are 16B b128; mode-0 stores are vectorized [N,M].
__global__ __launch_bounds__(256)
void gemm_bf16_kernel(const __bf16* __restrict__ A,    // [M, Kd]
                      const __bf16* __restrict__ Bm,   // [B, N, Kd]
                      float*  outF, __bf16* outB,
                      const float* __restrict__ bias,  // [M] or null
                      const float* __restrict__ resid, // or null
                      int M, int N, int Kd, float scale,
                      int relu, int residT, int outT) {
    const int lane = threadIdx.x & 31;
    const int wave = threadIdx.x >> 5;
    const int wm = wave >> 2;            // 0..1
    const int wn = wave & 3;             // 0..3
    const int n   = lane & 15;
    const int hw  = lane >> 4;
    const int kbA = hw * 8;
    const int kbB = hw * 8;
    const int mb  = hw * 8;

    const int M0 = blockIdx.y * 64 + wm * 32;
    const int N0 = blockIdx.x * 256 + wn * 64;
    const int b  = blockIdx.z;
    const __bf16* Bb = Bm + (size_t)b * N * Kd;

    const v8f zero = {0.f,0.f,0.f,0.f,0.f,0.f,0.f,0.f};
    v8f acc[2][4];
    #pragma unroll
    for (int u = 0; u < 2; ++u)
        #pragma unroll
        for (int t = 0; t < 4; ++t) acc[u][t] = zero;

    for (int K0 = 0; K0 < Kd; K0 += 32) {
        AFrag a[2];
        #pragma unroll
        for (int u = 0; u < 2; ++u) {
            const __bf16* ap = A + (size_t)(M0 + u * 16 + n) * Kd + K0 + kbA;
            a[u].u[0] = *(const uint4*)(ap);
            a[u].u[1] = *(const uint4*)(ap + 16);
        }
        #pragma unroll
        for (int t = 0; t < 4; ++t) {
            AFrag bf;
            const __bf16* bp = Bb + (size_t)(N0 + t * 16 + n) * Kd + K0 + kbB;
            bf.u[0] = *(const uint4*)(bp);
            bf.u[1] = *(const uint4*)(bp + 16);
            acc[0][t] = wmma_bf16(a[0].v, bf.v, acc[0][t]);
            acc[1][t] = wmma_bf16(a[1].v, bf.v, acc[1][t]);
        }
    }

    const size_t cb = (size_t)b * M * N;
    #pragma unroll
    for (int u = 0; u < 2; ++u) {
        #pragma unroll
        for (int t = 0; t < 4; ++t) {
            const int m0 = M0 + u * 16 + mb;
            const int nn = N0 + t * 16 + n;
            float v[8];
            #pragma unroll
            for (int r = 0; r < 8; ++r) {
                float val = acc[u][t][r] * scale;
                if (bias)  val += bias[m0 + r];
                if (resid) val += residT ? resid[cb + (size_t)(m0 + r) * N + nn]
                                         : resid[cb + (size_t)nn * M + m0 + r];
                if (relu)  val = fmaxf(val, 0.f);
                v[r] = val;
            }
            if (!outT) {                       // [N, M]: 8 consecutive M per lane
                const size_t o = cb + (size_t)nn * M + m0;
                if (outF) {
                    float4 lo = {v[0], v[1], v[2], v[3]};
                    float4 hi = {v[4], v[5], v[6], v[7]};
                    *(float4*)&outF[o]     = lo;
                    *(float4*)&outF[o + 4] = hi;
                }
                if (outB) {
                    Pack8 p;
                    #pragma unroll
                    for (int r = 0; r < 8; ++r) p.h[r] = (__bf16)v[r];
                    *(uint4*)&outB[o] = p.u;
                }
            } else {                           // [M, N] scatter (final output / V)
                #pragma unroll
                for (int r = 0; r < 8; ++r) {
                    const size_t o = cb + (size_t)(m0 + r) * N + nn;
                    if (outF) outF[o] = v[r];
                    if (outB) outB[o] = (__bf16)v[r];
                }
            }
        }
    }
}

// ---------------------------------------------------------------- flash attention
// One wave per (b, h, 16-query block). Q pre-scaled by 1/sqrt(DK).
// Q,K,Ho: bf16 [B, L, D] (K contiguous);  V: bf16 [B, D, L] (j contiguous).
__global__ __launch_bounds__(32)
void flash_attn_kernel(const __bf16* __restrict__ Q,
                       const __bf16* __restrict__ K,
                       const __bf16* __restrict__ V,
                       __bf16* __restrict__ Ho) {
    __shared__ __align__(16) float  Sh[16][32];
    __shared__ __align__(16) __bf16 Ph[16][32];
    __shared__ float alphaSh[16];
    __shared__ float sumSh[16];

    const int lane = threadIdx.x & 31;
    const int n   = lane & 15;
    const int hw  = lane >> 4;
    const int kbA = hw * 8;
    const int kbB = hw * 8;
    const int mb  = hw * 8;

    const int b  = blockIdx.z;
    const int h  = blockIdx.y;
    const int I0 = blockIdx.x * 16;
    const size_t rowBase = (size_t)b * LSEQ;           // row index base in [B*L, D]
    const int    cBase   = h * DKH;                    // channel base
    const size_t vBase   = ((size_t)b * DCH + cBase) * LSEQ;

    // Q^T A-fragments: [16 i x 64 dk] -> 2 K-steps, each two b128 loads
    AFrag qa[2];
    #pragma unroll
    for (int s = 0; s < 2; ++s) {
        const __bf16* qp = Q + (rowBase + I0 + n) * DCH + cBase + s * 32 + kbA;
        qa[s].u[0] = *(const uint4*)(qp);
        qa[s].u[1] = *(const uint4*)(qp + 16);
    }

    const v8f zero = {0.f,0.f,0.f,0.f,0.f,0.f,0.f,0.f};
    v8f o[4] = {zero, zero, zero, zero};
    float mrow = -INFINITY;
    float srow = 0.f;

    for (int j0 = 0; j0 < LSEQ; j0 += 32) {
        if (j0 + 32 < LSEQ) {   // prefetch next chunk (global_prefetch_b8)
            __builtin_prefetch(K + (rowBase + j0 + 32 + n) * DCH + cBase, 0, 1);
            __builtin_prefetch(V + vBase + (size_t)n * LSEQ + j0 + 32, 0, 1);
        }
        // ---- scores S[16 i x 32 j] = Q^T * K : all operands contiguous b128
        v8f s0 = zero, s1 = zero;
        #pragma unroll
        for (int s = 0; s < 2; ++s) {
            AFrag kb0, kb1;
            const __bf16* k0p = K + (rowBase + j0 + n)      * DCH + cBase + s * 32 + kbB;
            const __bf16* k1p = K + (rowBase + j0 + 16 + n) * DCH + cBase + s * 32 + kbB;
            kb0.u[0] = *(const uint4*)(k0p); kb0.u[1] = *(const uint4*)(k0p + 16);
            kb1.u[0] = *(const uint4*)(k1p); kb1.u[1] = *(const uint4*)(k1p + 16);
            s0 = wmma_bf16(qa[s].v, kb0.v, s0);
            s1 = wmma_bf16(qa[s].v, kb1.v, s1);
        }
        // spill C-layout scores to LDS (transpose for softmax + P fragment)
        #pragma unroll
        for (int r = 0; r < 8; ++r) {
            Sh[mb + r][n]      = s0[r];
            Sh[mb + r][16 + n] = s1[r];
        }
        __syncthreads();

        // ---- online softmax: lanes 0..15 own one query row each
        if (lane < 16) {
            float mx = mrow;
            #pragma unroll 8
            for (int j = 0; j < 32; ++j) mx = fmaxf(mx, Sh[lane][j]);
            const float alpha = __expf(mrow - mx);
            float ssum = srow * alpha;
            #pragma unroll 8
            for (int j = 0; j < 32; ++j) {
                const float p = __expf(Sh[lane][j] - mx);
                ssum += p;
                Ph[lane][j] = (__bf16)p;
            }
            mrow = mx;
            srow = ssum;
            alphaSh[lane] = alpha;
        }
        __syncthreads();

        // ---- rescale running output
        #pragma unroll
        for (int r = 0; r < 8; ++r) {
            const float al = alphaSh[mb + r];
            #pragma unroll
            for (int t = 0; t < 4; ++t) o[t][r] *= al;
        }

        // ---- O += P (A-layout from LDS) x V^T (contiguous b128 from global)
        AFrag pa;
        pa.u[0] = *(const uint4*)&Ph[n][kbA];
        pa.u[1] = *(const uint4*)&Ph[n][16 + kbA];
        #pragma unroll
        for (int t = 0; t < 4; ++t) {
            AFrag vb;
            const __bf16* vp = V + vBase + (size_t)(t * 16 + n) * LSEQ + j0 + kbB;
            vb.u[0] = *(const uint4*)(vp);
            vb.u[1] = *(const uint4*)(vp + 16);
            o[t] = wmma_bf16(pa.v, vb.v, o[t]);
        }
        __syncthreads();
    }

    if (lane < 16) sumSh[lane] = srow;
    __syncthreads();

    // Ho[b, i, h*64+dk] (= [B, L, D], B-operand layout for the Wh GEMM)
    #pragma unroll
    for (int t = 0; t < 4; ++t) {
        #pragma unroll
        for (int r = 0; r < 8; ++r) {
            const int m = mb + r;
            const float val = o[t][r] / sumSh[m];
            Ho[(rowBase + I0 + m) * DCH + cBase + t * 16 + n] = (__bf16)val;
        }
    }
}

// ---------------------------------------------------------------- batchnorm
// stats on [B, L, D] layout: per-block partials (coalesced), deterministic reduce
__global__ __launch_bounds__(256)
void bn_stats_T_kernel(const float* __restrict__ y,   // [B*L, D]
                       float* __restrict__ part) {    // [grid][1024]
    float s0 = 0.f, q0 = 0.f, s1 = 0.f, q1 = 0.f;
    const int c0 = threadIdx.x, c1 = threadIdx.x + 256;
    for (int row = blockIdx.x; row < BSZ * LSEQ; row += gridDim.x) {
        const float* r = y + (size_t)row * DCH;
        const float a = r[c0], bb = r[c1];
        s0 += a;  q0 += a * a;
        s1 += bb; q1 += bb * bb;
    }
    float* p = part + (size_t)blockIdx.x * 1024;
    p[c0] = s0;  p[512 + c0] = q0;
    p[c1] = s1;  p[512 + c1] = q1;
}

__global__ void bn_finalize_kernel(const float* __restrict__ part, int nparts,
                                   float* __restrict__ mean, float* __restrict__ rstd) {
    const int c = blockIdx.x * blockDim.x + threadIdx.x;
    if (c >= DCH) return;
    float s = 0.f, q = 0.f;
    for (int k = 0; k < nparts; ++k) {
        s += part[(size_t)k * 1024 + c];
        q += part[(size_t)k * 1024 + 512 + c];
    }
    const float inv = 1.f / (float)(BSZ * LSEQ);
    const float mu  = s * inv;
    const float var = q * inv - mu * mu;
    mean[c] = mu;
    rstd[c] = rsqrtf(var + 1e-5f);
}

// stats on [B, D, L] layout (coalesced block-per-channel) — for BN2
__global__ __launch_bounds__(256)
void bn_stats_kernel(const float* __restrict__ y, float* __restrict__ mean,
                     float* __restrict__ rstd) {
    const int c = blockIdx.x;
    const int tid = threadIdx.x;
    float s = 0.f, s2 = 0.f;
    for (int i = tid; i < BSZ * LSEQ; i += 256) {
        const int bb = i / LSEQ;
        const int l  = i - bb * LSEQ;
        const float v = y[((size_t)bb * DCH + c) * LSEQ + l];
        s += v; s2 += v * v;
    }
    __shared__ float sh[256], sh2[256];
    sh[tid] = s; sh2[tid] = s2;
    __syncthreads();
    for (int o = 128; o > 0; o >>= 1) {
        if (tid < o) { sh[tid] += sh[tid + o]; sh2[tid] += sh2[tid + o]; }
        __syncthreads();
    }
    if (tid == 0) {
        const float inv = 1.f / (float)(BSZ * LSEQ);
        const float mu  = sh[0] * inv;
        const float var = sh2[0] * inv - mu * mu;
        mean[c] = mu;
        rstd[c] = rsqrtf(var + 1e-5f);
    }
}

// normalize in place; channel = (i / div) % D  (div=1 for [L,D], div=L for [D,L])
__global__ void bn_norm_kernel(float* __restrict__ y,
                               const float* __restrict__ mean,
                               const float* __restrict__ rstd,
                               const float* __restrict__ g,
                               const float* __restrict__ be,
                               __bf16* outB, size_t total, int div) {
    for (size_t i = (size_t)blockIdx.x * blockDim.x + threadIdx.x; i < total;
         i += (size_t)gridDim.x * blockDim.x) {
        const int c = (int)((i / div) % DCH);
        const float v = (y[i] - mean[c]) * rstd[c] * g[c] + be[c];
        y[i] = v;
        if (outB) outB[i] = (__bf16)v;
    }
}

// ---------------------------------------------------------------- launch
extern "C" void kernel_launch(void* const* d_in, const int* in_sizes, int n_in,
                              void* d_out, int out_size, void* d_ws, size_t ws_size,
                              hipStream_t stream) {
    (void)in_sizes; (void)n_in; (void)out_size; (void)ws_size;
    const float* x   = (const float*)d_in[0];
    const float* Wq  = (const float*)d_in[1];
    const float* Wk  = (const float*)d_in[2];
    const float* Wv  = (const float*)d_in[3];
    const float* Wh  = (const float*)d_in[4];
    const float* bh  = (const float*)d_in[5];
    const float* W1  = (const float*)d_in[6];
    const float* b1  = (const float*)d_in[7];
    const float* W2  = (const float*)d_in[8];
    const float* b2  = (const float*)d_in[9];
    const float* g1  = (const float*)d_in[10];
    const float* be1 = (const float*)d_in[11];
    const float* g2  = (const float*)d_in[12];
    const float* be2 = (const float*)d_in[13];
    float* out = (float*)d_out;

    const size_t NX  = (size_t)BSZ * DCH * LSEQ;
    const size_t NW  = (size_t)DCH * DCH;
    const size_t NW1 = (size_t)HIDF * DCH;
    const size_t NH  = (size_t)BSZ * HIDF * LSEQ;
    const int NPART = 64;

    char* ws = (char*)d_ws;
    size_t off = 0;
    auto alloc = [&](size_t bytes) -> void* {
        void* p = ws + off;
        off = (off + bytes + 255) & ~(size_t)255;
        return p;
    };
    float*  xT   = (float*)alloc(NX * 4);    // x transposed [B, L, D] f32
    __bf16* xbT  = (__bf16*)alloc(NX * 2);   // x transposed bf16
    __bf16* Wqb  = (__bf16*)alloc(NW * 2);
    __bf16* Wkb  = (__bf16*)alloc(NW * 2);
    __bf16* Wvb  = (__bf16*)alloc(NW * 2);
    __bf16* Whb  = (__bf16*)alloc(NW * 2);
    __bf16* W1b  = (__bf16*)alloc(NW1 * 2);
    __bf16* W2b  = (__bf16*)alloc(NW1 * 2);
    __bf16* Qbf  = (__bf16*)alloc(NX * 2);   // [B, L, D]
    __bf16* Kbf  = (__bf16*)alloc(NX * 2);   // [B, L, D]
    __bf16* Vbf  = (__bf16*)alloc(NX * 2);   // [B, D, L]
    __bf16* Hobf = (__bf16*)alloc(NX * 2);   // [B, L, D]
    float*  y1   = (float*)alloc(NX * 4);    // [B, L, D]
    __bf16* o1b  = (__bf16*)alloc(NX * 2);   // [B, L, D]
    __bf16* hbf  = (__bf16*)alloc(NH * 2);   // [B, L, HID]
    float*  part  = (float*)alloc((size_t)NPART * 1024 * 4);
    float*  mean1 = (float*)alloc(DCH * 4);
    float*  rstd1 = (float*)alloc(DCH * 4);
    float*  mean2 = (float*)alloc(DCH * 4);
    float*  rstd2 = (float*)alloc(DCH * 4);

    // 1) transpose+convert x; convert weights
    transpose_cvt_kernel<<<dim3(LSEQ / 32, DCH / 32, BSZ), dim3(32, 8), 0, stream>>>(x, xT, xbT);
    cvt_bf16_kernel<<<512,  256, 0, stream>>>(Wq, Wqb, NW);
    cvt_bf16_kernel<<<512,  256, 0, stream>>>(Wk, Wkb, NW);
    cvt_bf16_kernel<<<512,  256, 0, stream>>>(Wv, Wvb, NW);
    cvt_bf16_kernel<<<512,  256, 0, stream>>>(Wh, Whb, NW);
    cvt_bf16_kernel<<<1024, 256, 0, stream>>>(W1, W1b, NW1);
    cvt_bf16_kernel<<<1024, 256, 0, stream>>>(W2, W2b, NW1);

    const float qscale = 0.125f;    // 1/sqrt(64)
    const dim3 gD(LSEQ / 256, DCH / 64, BSZ);
    // 2) projections: Q,K -> [L,D]; V -> [D,L] (scatter epilogue)
    gemm_bf16_kernel<<<gD, 256, 0, stream>>>(Wqb, xbT, nullptr, Qbf, nullptr, nullptr,
                                             DCH, LSEQ, DCH, qscale, 0, 0, 0);
    gemm_bf16_kernel<<<gD, 256, 0, stream>>>(Wkb, xbT, nullptr, Kbf, nullptr, nullptr,
                                             DCH, LSEQ, DCH, 1.f, 0, 0, 0);
    gemm_bf16_kernel<<<gD, 256, 0, stream>>>(Wvb, xbT, nullptr, Vbf, nullptr, nullptr,
                                             DCH, LSEQ, DCH, 1.f, 0, 0, 1);

    // 3) flash attention -> Ho [B, L, D]
    flash_attn_kernel<<<dim3(LSEQ / 16, HHEAD, BSZ), 32, 0, stream>>>(Qbf, Kbf, Vbf, Hobf);

    // 4) y1 = Wh*Ho + bh + x   (out [L,D], residual xT [L,D])
    gemm_bf16_kernel<<<gD, 256, 0, stream>>>(Whb, Hobf, y1, nullptr, bh, xT,
                                             DCH, LSEQ, DCH, 1.f, 0, 0, 0);

    // 5) batchnorm 1 in place + bf16 copy
    bn_stats_T_kernel<<<NPART, 256, 0, stream>>>(y1, part);
    bn_finalize_kernel<<<2, 256, 0, stream>>>(part, NPART, mean1, rstd1);
    bn_norm_kernel<<<4096, 256, 0, stream>>>(y1, mean1, rstd1, g1, be1, o1b, NX, 1);

    // 6) FFN1: h = relu(W1*out1 + b1) -> [L, HID]
    gemm_bf16_kernel<<<dim3(LSEQ / 256, HIDF / 64, BSZ), 256, 0, stream>>>(
        W1b, o1b, nullptr, hbf, b1, nullptr, HIDF, LSEQ, DCH, 1.f, 1, 0, 0);

    // 7) FFN2 + bias + residual(y1 [L,D]) -> d_out in [B, D, L] (scatter epilogue)
    gemm_bf16_kernel<<<gD, 256, 0, stream>>>(W2b, hbf, out, nullptr, b2, y1,
                                             DCH, LSEQ, HIDF, 1.f, 0, 0, 1);

    // 8) batchnorm 2 in place on d_out ([D,L] layout, coalesced)
    bn_stats_kernel<<<DCH, 256, 0, stream>>>(out, mean2, rstd2);
    bn_norm_kernel<<<4096, 256, 0, stream>>>(out, mean2, rstd2, g2, be2, nullptr, NX, LSEQ);
}
